// Simple_32091995636276
// MI455X (gfx1250) — matrix operand order repaired and checked by reading, not compiled, not run on hardware
//
#include <hip/hip_runtime.h>
#include <hip/hip_bf16.h>
#include <math.h>

// ---------------------------------------------------------------------------
// CDNA5 (gfx1250) implementation of the FPN + ROI-correlation reference.
// All convolutions run as implicit GEMM through v_wmma_f32_16x16x32_bf16,
// with 2x2 fragment tiling per wave (4 WMMAs per K-chunk per wave).
// ---------------------------------------------------------------------------

typedef __attribute__((ext_vector_type(16))) __bf16 v16bf;
typedef __attribute__((ext_vector_type(8)))  float  v8f;

#define BK  32  // K chunk per WMMA (bf16)
#define SBS 34  // LDS row stride in bf16 (17 dwords, coprime with 64 banks)

// ------------------------- WMMA implicit-GEMM conv -------------------------
// out[b, co, y, x] = act( (sum_{ci,ky,kx} in[b,ci,y+ky-pad,x+kx-pad]*w[co,ci,ky,kx]
//                          + bias[co]) * (scale[co]*kscale) + shift[co] ) * postm + postb
// Block: 256 threads = 8 waves. Block tile M=32 (C_out) x N=256 spatial.
// Each wave computes a 32x32 patch = 2x2 WMMA fragments.
__global__ __launch_bounds__(256) void conv_wmma_k(
    const float* __restrict__ in, const float* __restrict__ wgt,
    const float* __restrict__ bias, const float* __restrict__ scale,
    const float* __restrict__ shift, float* __restrict__ out,
    int Cin, int Cout, int H, int W, int KH, int KW, int pad,
    long inB, long outB, long wB, int mode, float kscale, float postm, float postb)
{
  __shared__ __bf16 sA[32 * SBS];
  __shared__ __bf16 sB[256 * SBS];
  const int tid  = threadIdx.x;
  const int lane = tid & 31;
  const int wave = tid >> 5;
  const int hi   = lane >> 4;      // lane half (selects k = 16*hi + e, contiguous order)
  const int mrow = lane & 15;
  const int b    = blockIdx.z;
  const int co0  = blockIdx.x * 32;
  const int HW   = H * W;
  const int n0   = blockIdx.y * 256;
  const float* inb = in + (size_t)b * inB;
  const float* wgb = wgt + (size_t)b * wB;

  // This thread permanently stages spatial column n0+tid of the B tile.
  const int  nst    = n0 + tid;
  const bool nvalid = nst < HW;
  const int  yst    = nvalid ? nst / W : 0;
  const int  xst    = nvalid ? nst % W : 0;

  v8f acc00 = {0.f,0.f,0.f,0.f,0.f,0.f,0.f,0.f};
  v8f acc01 = acc00, acc10 = acc00, acc11 = acc00;

  for (int ky = 0; ky < KH; ++ky)
    for (int kx = 0; kx < KW; ++kx) {
      const int yb = yst + ky - pad;
      const int xb = xst + kx - pad;
      const bool inside = nvalid && yb >= 0 && yb < H && xb >= 0 && xb < W;
      const float* src0 = &inb[(size_t)yb * W + xb];  // + ci*HW per channel

      for (int c0 = 0; c0 < Cin; c0 += BK) {
        // ---- stage A: 32 (co) x 32 (ci) weights ----
#pragma unroll
        for (int it = 0; it < 4; ++it) {
          int idx = tid + it * 256;
          int m = idx >> 5, k = idx & 31;
          int co = co0 + m, ci = c0 + k;
          float v = 0.f;
          if (co < Cout && ci < Cin)
            v = wgb[(((size_t)co * Cin + ci) * KH + ky) * KW + kx];
          sA[m * SBS + k] = (__bf16)v;
        }
        // ---- stage B: 256 (n) x 32 (ci); coalesced channel stream ----
        if (inside && c0 + BK < Cin)  // speculative prefetch of next chunk
          __builtin_prefetch(src0 + (size_t)(c0 + BK) * HW, 0, 1);
#pragma unroll 8
        for (int k = 0; k < BK; ++k) {
          int ci = c0 + k;
          float v = 0.f;
          if (inside && ci < Cin) v = src0[(size_t)ci * HW];
          sB[tid * SBS + k] = (__bf16)v;
        }
        __syncthreads();

        // ---- build fragments: contiguous k = 16*hi + e for A and B ----
        v16bf a0, a1, b0, b1;
#pragma unroll
        for (int r = 0; r < 8; ++r) {
#pragma unroll
          for (int t = 0; t < 2; ++t) {
            int e  = 2 * r + t;
            int kA = hi * 16 + 2 * r + t;
            a0[e] = sA[mrow * SBS + kA];
            a1[e] = sA[(16 + mrow) * SBS + kA];
            int kB = hi * 16 + 2 * r + t;
            b0[e] = sB[(wave * 32 + mrow) * SBS + kB];
            b1[e] = sB[(wave * 32 + 16 + mrow) * SBS + kB];
          }
        }
        acc00 = __builtin_amdgcn_wmma_f32_16x16x32_bf16(
            false, a0, false, b0, (short)0, acc00, false, false);
        acc01 = __builtin_amdgcn_wmma_f32_16x16x32_bf16(
            false, a0, false, b1, (short)0, acc01, false, false);
        acc10 = __builtin_amdgcn_wmma_f32_16x16x32_bf16(
            false, a1, false, b0, (short)0, acc10, false, false);
        acc11 = __builtin_amdgcn_wmma_f32_16x16x32_bf16(
            false, a1, false, b1, (short)0, acc11, false, false);
        __syncthreads();
      }
    }

  // ---- epilogue: C/D layout -> row m = r + hi*8, col n = lane&15 ----
#pragma unroll
  for (int mi = 0; mi < 2; ++mi)
#pragma unroll
    for (int ni = 0; ni < 2; ++ni) {
      v8f acc = (mi == 0) ? (ni == 0 ? acc00 : acc01)
                          : (ni == 0 ? acc10 : acc11);
      int n = n0 + wave * 32 + ni * 16 + mrow;
      if (n < HW) {
#pragma unroll
        for (int r = 0; r < 8; ++r) {
          int co = co0 + mi * 16 + r + hi * 8;
          if (co < Cout) {
            float y = acc[r];
            if (bias)  y += bias[co];
            if (scale) y = y * (scale[co] * kscale) + shift[co];
            if (mode == 1)      y = fmaxf(y, 0.f);
            else if (mode == 2) y = 1.f / (1.f + __expf(-y));
            y = y * postm + postb;
            out[(size_t)b * outB + (size_t)co * HW + n] = y;
          }
        }
      }
    }
}

// ------------------------- bilinear resize (align_corners) -----------------
__global__ void resize_ac_k(const float* __restrict__ in,
                            const float* __restrict__ add,
                            float* __restrict__ out, int NB, int C,
                            int ih, int iw, int oh, int ow,
                            long inB, long addB, long outB)
{
  long total = (long)NB * C * oh * ow;
  long idx = (long)blockIdx.x * blockDim.x + threadIdx.x;
  if (idx >= total) return;
  int x = (int)(idx % ow); long t = idx / ow;
  int y = (int)(t % oh);   t /= oh;
  int c = (int)(t % C);
  int n = (int)(t / C);
  float fy = (oh > 1) ? (float)(ih - 1) / (float)(oh - 1) : 0.f;
  float fx = (ow > 1) ? (float)(iw - 1) / (float)(ow - 1) : 0.f;
  float cy = y * fy, cx = x * fx;
  int y0 = (int)floorf(cy); y0 = y0 < 0 ? 0 : (y0 > ih - 1 ? ih - 1 : y0);
  int x0 = (int)floorf(cx); x0 = x0 < 0 ? 0 : (x0 > iw - 1 ? iw - 1 : x0);
  int y1 = y0 + 1 > ih - 1 ? ih - 1 : y0 + 1;
  int x1 = x0 + 1 > iw - 1 ? iw - 1 : x0 + 1;
  float gy = cy - (float)y0, gx = cx - (float)x0;
  const float* p = in + (size_t)n * inB + (size_t)c * ih * iw;
  float v00 = p[y0 * iw + x0], v01 = p[y0 * iw + x1];
  float v10 = p[y1 * iw + x0], v11 = p[y1 * iw + x1];
  float v = (v00 * (1.f - gx) + v01 * gx) * (1.f - gy) +
            (v10 * (1.f - gx) + v11 * gx) * gy;
  if (add) v += add[(size_t)n * addB + (size_t)c * oh * ow + y * ow + x];
  out[(size_t)n * outB + (size_t)c * oh * ow + y * ow + x] = v;
}

// ------------------------- ROI pooling (per box) ----------------------------
// One block per (batch, box). Resize crop -> center pad -> sum-normalize.
__global__ __launch_bounds__(256) void roi_k(
    const float* __restrict__ feat, const float* __restrict__ xex,
    float* __restrict__ ze, int rs, int C, int H, int W, int M)
{
  int i = blockIdx.x / M, j = blockIdx.x % M;
  const float* bp = xex + (size_t)(i * M + j) * 4;
  int x_min = (int)(bp[0] / 8.0f);
  int y_min = (int)(bp[1] / 8.0f);
  int bw    = (int)(bp[2] / 8.0f);
  int bh    = (int)(bp[3] / 8.0f);
  int rhin = bh + 1, rwin = bw + 1;
  int mx = bh > bw ? bh : bw;
  float scl = (float)mx / (float)rs;
  int nh = (int)((float)bh / scl); if (nh < 1) nh = 1; if (nh > rs) nh = rs;
  int nw = (int)((float)bw / scl); if (nw < 1) nw = 1; if (nw > rs) nw = rs;
  int tp = (rs - nh) / 2, lp = (rs - nw) / 2;
  const float* fb = feat + (size_t)i * C * H * W;
  float fyfac = (nh > 1) ? (float)(rhin - 1) / (float)(nh - 1) : 0.f;
  float fxfac = (nw > 1) ? (float)(rwin - 1) / (float)(nw - 1) : 0.f;

  auto sample = [&](int c, int ry, int rx) -> float {
    float cy = ry * fyfac, cx = rx * fxfac;
    int y0 = (int)cy; if (y0 > rhin - 1) y0 = rhin - 1;
    int x0 = (int)cx; if (x0 > rwin - 1) x0 = rwin - 1;
    int y1 = y0 + 1 > rhin - 1 ? rhin - 1 : y0 + 1;
    int x1 = x0 + 1 > rwin - 1 ? rwin - 1 : x0 + 1;
    float gy = cy - (float)y0, gx = cx - (float)x0;
    const float* p = fb + (size_t)c * H * W;
    float v00 = p[(y_min + y0) * W + x_min + x0];
    float v01 = p[(y_min + y0) * W + x_min + x1];
    float v10 = p[(y_min + y1) * W + x_min + x0];
    float v11 = p[(y_min + y1) * W + x_min + x1];
    return (v00 * (1.f - gx) + v01 * gx) * (1.f - gy) +
           (v10 * (1.f - gx) + v11 * gx) * gy;
  };

  // pass 1: total sum of resized region (padding contributes 0)
  float lsum = 0.f;
  int tot = C * nh * nw;
  for (int idx = threadIdx.x; idx < tot; idx += blockDim.x) {
    int rx = idx % nw; int t = idx / nw;
    int ry = t % nh;   int c = t / nh;
    lsum += sample(c, ry, rx);
  }
  __shared__ float red[256];
  red[threadIdx.x] = lsum;
  __syncthreads();
  for (int o = 128; o > 0; o >>= 1) {
    if (threadIdx.x < o) red[threadIdx.x] += red[threadIdx.x + o];
    __syncthreads();
  }
  float inv = 1.f / (1e-6f + red[0]);

  // pass 2: write normalized, center-padded output
  float* zb = ze + (size_t)(i * M + j) * C * rs * rs;
  int tot2 = C * rs * rs;
  for (int idx = threadIdx.x; idx < tot2; idx += blockDim.x) {
    int xx = idx % rs; int t = idx / rs;
    int yy = t % rs;   int c = t / rs;
    int ry = yy - tp, rx = xx - lp;
    float v = 0.f;
    if (ry >= 0 && ry < nh && rx >= 0 && rx < nw)
      v = sample(c, ry, rx) * inv;
    zb[idx] = v;
  }
}

// ------------------- build correlation filters (aug + pad) -----------------
// wall[b, s*20+m, c, ky, kx]; m: q=m/5 flip selector, j=m%5 box.
__global__ void wall_k(const float* __restrict__ ze0,
                       const float* __restrict__ ze1,
                       const float* __restrict__ ze2,
                       float* __restrict__ wall, long total)
{
  long idx = (long)blockIdx.x * blockDim.x + threadIdx.x;
  if (idx >= total) return;
  int kx = (int)(idx % 13); long t = idx / 13;
  int ky = (int)(t % 13);   t /= 13;
  int c  = (int)(t % 256);  t /= 256;
  int oc = (int)(t % 60);
  int b  = (int)(t / 60);
  int s = oc / 20, m = oc % 20;
  int q = m / 5, j = m % 5;
  int rs = (s == 0) ? 4 : (s == 1) ? 7 : 13;
  int ph = (13 - rs) / 2;
  float v = 0.f;
  int iy = ky - ph, ix = kx - ph;
  if (iy >= 0 && iy < rs && ix >= 0 && ix < rs) {
    int ys = (q >= 2) ? (rs - 1 - iy) : iy;
    int xs = (q & 1) ? (rs - 1 - ix) : ix;
    const float* ze = (s == 0) ? ze0 : (s == 1) ? ze1 : ze2;
    v = ze[(((size_t)(b * 5 + j) * 256 + c) * rs + ys) * rs + xs];
  }
  wall[idx] = v;
}

// ------------------------------ small kernels ------------------------------
__global__ void fill_k(float* __restrict__ p, float v, long n) {
  long i = (long)blockIdx.x * blockDim.x + threadIdx.x;
  if (i < n) p[i] = v;
}
// replicate conv_map.at[:, s:s+20].set(res): h channel base offset = s
__global__ void copy_slice_k(const float* __restrict__ res,
                             float* __restrict__ h, int s, long total) {
  long i = (long)blockIdx.x * blockDim.x + threadIdx.x;
  if (i >= total) return;
  int p = (int)(i % 4096); long t = i / 4096;
  int c = (int)(t % 20);
  int b = (int)(t / 20);
  h[((size_t)b * 316 + s + c) * 4096 + p] = res[((size_t)b * 20 + c) * 4096 + p];
}
__global__ void mean_k(const float* __restrict__ h, float* __restrict__ att,
                       long total) {
  long i = (long)blockIdx.x * blockDim.x + threadIdx.x;
  if (i >= total) return;
  int p = (int)(i % 4096);
  int b = (int)(i / 4096);
  float s = 0.f;
  for (int c = 0; c < 60; ++c) s += h[((size_t)b * 316 + c) * 4096 + p];
  att[(size_t)b * 4096 + p] = s * (1.f / 60.f);
}
__global__ void attmul_k(const float* __restrict__ z1,
                         const float* __restrict__ att,
                         float* __restrict__ h, long total) {
  long i = (long)blockIdx.x * blockDim.x + threadIdx.x;
  if (i >= total) return;
  int p = (int)(i % 4096); long t = i / 4096;
  int c = (int)(t % 256);
  int b = (int)(t / 256);
  h[((size_t)b * 316 + 60 + c) * 4096 + p] =
      z1[((size_t)b * 256 + c) * 4096 + p] * att[(size_t)b * 4096 + p];
}

// ------------------------------ host helpers -------------------------------
static inline int nblk(long n, int t) { return (int)((n + t - 1) / t); }

static void conv(hipStream_t s, const float* in, const float* w,
                 const float* bias, const float* scale, const float* shift,
                 float* out, int Cin, int Cout, int H, int W, int KH, int KW,
                 int pad, long inB, long outB, long wB, int NB, int mode,
                 float kscale = 1.f, float postm = 1.f, float postb = 0.f) {
  dim3 g((Cout + 31) / 32, (H * W + 255) / 256, NB);
  conv_wmma_k<<<g, 256, 0, s>>>(in, w, bias, scale, shift, out, Cin, Cout, H,
                                W, KH, KW, pad, inB, outB, wB, mode, kscale,
                                postm, postb);
}

extern "C" void kernel_launch(void* const* d_in, const int* in_sizes, int n_in,
                              void* d_out, int out_size, void* d_ws,
                              size_t ws_size, hipStream_t stream) {
  const float* x0 = (const float*)d_in[0];
  const float* x1 = (const float*)d_in[1];
  const float* x2 = (const float*)d_in[2];
  const float* x3 = (const float*)d_in[3];
  const float* xex = (const float*)d_in[4];
  const float* c1w0 = (const float*)d_in[5];  const float* c1b0 = (const float*)d_in[6];
  const float* c1w1 = (const float*)d_in[7];  const float* c1b1 = (const float*)d_in[8];
  const float* c1w2 = (const float*)d_in[9];  const float* c1b2 = (const float*)d_in[10];
  const float* smw = (const float*)d_in[11];  const float* smb = (const float*)d_in[12];
  const float* fww = (const float*)d_in[13];
  const float* bng = (const float*)d_in[14];  const float* bnb = (const float*)d_in[15];
  const float* llw1 = (const float*)d_in[16]; const float* llb1 = (const float*)d_in[17];
  const float* llw2 = (const float*)d_in[18]; const float* llb2 = (const float*)d_in[19];
  const float* llw3 = (const float*)d_in[20]; const float* llb3 = (const float*)d_in[21];
  const float* ofw1 = (const float*)d_in[22]; const float* ofb1 = (const float*)d_in[23];
  const float* ofw2 = (const float*)d_in[24]; const float* ofb2 = (const float*)d_in[25];
  const float* ofw3 = (const float*)d_in[26]; const float* ofb3 = (const float*)d_in[27];
  float* ws = (float*)d_ws;
  float* out = (float*)d_out;

  // ---- workspace arena (elements). Phase 2 overlays dead phase-1 temps. ----
  const size_t F1 = 0,        F2 = 524288,  F3 = 655360;
  const size_t T3 = 688128,   S3 = 819200;
  const size_t T2 = 950272,   S2 = 1474560;
  const size_t T1 = 1998848;
  const size_t CAT = 4096000;              // (2,1024,64,64)
  const size_t Z1 = 12484608;              // (2,256,64,64) persists
  const size_t ZE0 = 0, ZE1 = 40960;       // phase-2 overlay
  const size_t WALL = 166400;              // (2,60,256,13,13)
  const size_t HB = 5358080;               // (2,316,64,64)
  const size_t RES = 7946752;              // (2,20,64,64)
  const size_t G1 = 8110592, G2 = 8634880;
  const size_t O1 = 9159168, O2 = 9683456;
  const size_t OUT1S = 10207744, OSM = 10215936;
  // d_out slices
  const size_t oOUT1 = 0, oOFF = 524288, oATT = 1572864, oZE2 = 1581056;

  // 1) lateral 1x1 convs to 256 ch
  conv(stream, x1, c1w0, c1b0, nullptr, nullptr, ws + F1, 512, 256, 32, 32, 1, 1, 0,
       512L * 1024, 256L * 1024, 0, 2, 0);
  conv(stream, x2, c1w1, c1b1, nullptr, nullptr, ws + F2, 1024, 256, 16, 16, 1, 1, 0,
       1024L * 256, 256L * 256, 0, 2, 0);
  conv(stream, x3, c1w2, c1b2, nullptr, nullptr, ws + F3, 2048, 256, 8, 8, 1, 1, 0,
       2048L * 64, 256L * 64, 0, 2, 0);

  // 2) top-down: upsample + add, then 3x3 smooth
  long n;
  n = 2L * 256 * 16 * 16;
  resize_ac_k<<<nblk(n, 256), 256, 0, stream>>>(ws + F3, ws + F2, ws + T3, 2, 256,
      8, 8, 16, 16, 256L * 64, 256L * 256, 256L * 256);
  conv(stream, ws + T3, smw, smb, nullptr, nullptr, ws + S3, 256, 256, 16, 16, 3, 3, 1,
       256L * 256, 256L * 256, 0, 2, 0);
  n = 2L * 256 * 32 * 32;
  resize_ac_k<<<nblk(n, 256), 256, 0, stream>>>(ws + F2, ws + F1, ws + T2, 2, 256,
      16, 16, 32, 32, 256L * 256, 256L * 1024, 256L * 1024);
  conv(stream, ws + T2, smw, smb, nullptr, nullptr, ws + S2, 256, 256, 32, 32, 3, 3, 1,
       256L * 1024, 256L * 1024, 0, 2, 0);
  n = 2L * 256 * 64 * 64;
  resize_ac_k<<<nblk(n, 256), 256, 0, stream>>>(ws + F1, x0, ws + T1, 2, 256,
      32, 32, 64, 64, 256L * 1024, 256L * 4096, 256L * 4096);
  conv(stream, ws + T1, smw, smb, nullptr, nullptr, ws + CAT, 256, 256, 64, 64, 3, 3, 1,
       256L * 4096, 1024L * 4096, 0, 2, 0);   // -> CAT channels 0..255

  // 3) upsample pyramid levels into CAT channels 256.. / 512.. / 768..
  n = 2L * 256 * 64 * 64;
  resize_ac_k<<<nblk(n, 256), 256, 0, stream>>>(ws + S2, nullptr,
      ws + CAT + 256L * 4096, 2, 256, 32, 32, 64, 64, 256L * 1024, 0, 1024L * 4096);
  resize_ac_k<<<nblk(n, 256), 256, 0, stream>>>(ws + S3, nullptr,
      ws + CAT + 512L * 4096, 2, 256, 16, 16, 64, 64, 256L * 256, 0, 1024L * 4096);
  resize_ac_k<<<nblk(n, 256), 256, 0, stream>>>(ws + F3, nullptr,
      ws + CAT + 768L * 4096, 2, 256, 8, 8, 64, 64, 256L * 64, 0, 1024L * 4096);

  // 4) fuse conv 1024->256 + folded BN-like scale + ReLU -> z1
  const float kbn = 1.0f / sqrtf(1.0f + 1e-5f);
  conv(stream, ws + CAT, fww, nullptr, bng, bnb, ws + Z1, 1024, 256, 64, 64, 3, 3, 1,
       1024L * 4096, 256L * 4096, 0, 2, 1, kbn);

  // 5) ROI pooling at 3 scales (scale 2 -> d_out z_e slice)
  roi_k<<<10, 256, 0, stream>>>(ws + Z1, xex, ws + ZE0, 4, 256, 64, 64, 5);
  roi_k<<<10, 256, 0, stream>>>(ws + Z1, xex, ws + ZE1, 7, 256, 64, 64, 5);
  roi_k<<<10, 256, 0, stream>>>(ws + Z1, xex, out + oZE2, 13, 256, 64, 64, 5);

  // 6) build augmented + padded correlation filters
  n = 2L * 60 * 256 * 13 * 13;
  wall_k<<<nblk(n, 256), 256, 0, stream>>>(ws + ZE0, ws + ZE1, out + oZE2,
                                           ws + WALL, n);

  // 7) zero h (conv_map channels 22..59 must stay zero)
  n = 2L * 316 * 4096;
  fill_k<<<nblk(n, 256), 256, 0, stream>>>(ws + HB, 0.f, n);

  // 8) correlation convs (per-batch 13x13 filters), overlapping slice writes
  for (int s = 0; s < 3; ++s) {
    conv(stream, ws + Z1, ws + WALL + (size_t)s * 20 * 256 * 169, nullptr, nullptr,
         nullptr, ws + RES, 256, 20, 64, 64, 13, 13, 6,
         256L * 4096, 20L * 4096, 60L * 256 * 169, 2, 0);
    long m = 2L * 20 * 4096;
    copy_slice_k<<<nblk(m, 256), 256, 0, stream>>>(ws + RES, ws + HB, s, m);
  }

  // 9) attention = mean over 60 channels -> d_out atten slice
  n = 2L * 4096;
  mean_k<<<nblk(n, 256), 256, 0, stream>>>(ws + HB, out + oATT, n);
  // 10) h[:,60:316] = z1 * atten
  n = 2L * 256 * 4096;
  attmul_k<<<nblk(n, 256), 256, 0, stream>>>(ws + Z1, out + oATT, ws + HB, n);

  // 11) ll head: 316->64->64->1 (relu)
  conv(stream, ws + HB, llw1, llb1, nullptr, nullptr, ws + G1, 316, 64, 64, 64, 3, 3, 1,
       316L * 4096, 64L * 4096, 0, 2, 1);
  conv(stream, ws + G1, llw2, llb2, nullptr, nullptr, ws + G2, 64, 64, 64, 64, 3, 3, 1,
       64L * 4096, 64L * 4096, 0, 2, 1);
  conv(stream, ws + G2, llw3, llb3, nullptr, nullptr, ws + OUT1S, 64, 1, 64, 64, 1, 1, 0,
       64L * 4096, 1L * 4096, 0, 2, 1);

  // 12) offset head: 256->64->64->2, sigmoid then affine 64*o - 32
  conv(stream, ws + Z1, ofw1, ofb1, nullptr, nullptr, ws + O1, 256, 64, 64, 64, 3, 3, 1,
       256L * 4096, 64L * 4096, 0, 2, 1);
  conv(stream, ws + O1, ofw2, ofb2, nullptr, nullptr, ws + O2, 64, 64, 64, 64, 3, 3, 1,
       64L * 4096, 64L * 4096, 0, 2, 1);
  conv(stream, ws + O2, ofw3, ofb3, nullptr, nullptr, ws + OSM, 64, 2, 64, 64, 1, 1, 0,
       64L * 4096, 2L * 4096, 0, 2, 2, 1.f, 64.f, -32.f);

  // 13) final align-corners upsamples to 512x512 into d_out
  n = 2L * 1 * 512 * 512;
  resize_ac_k<<<nblk(n, 256), 256, 0, stream>>>(ws + OUT1S, nullptr, out + oOUT1,
      2, 1, 64, 64, 512, 512, 1L * 4096, 0, 1L * 262144);
  n = 2L * 2 * 512 * 512;
  resize_ac_k<<<nblk(n, 256), 256, 0, stream>>>(ws + OSM, nullptr, out + oOFF,
      2, 2, 64, 64, 512, 512, 2L * 4096, 0, 2L * 262144);
}
